// HMM_18090402251289
// MI455X (gfx1250) — matrix operand order repaired and checked by reading, not compiled, not run on hardware
//
#include <hip/hip_runtime.h>
#include <hip/hip_bf16.h>

typedef __attribute__((ext_vector_type(16))) _Float16 v16h;
typedef __attribute__((ext_vector_type(8)))  _Float16 v8h;
typedef __attribute__((ext_vector_type(8)))  float    v8f;

#define B_TOT   4096
#define SEQLEN  512
#define W       64
#define NBSYM   10
#define NSTEPS  (SEQLEN - 1)   // 511 transition matrices
#define BM      16             // rows (batch elems) per block
#define THREADS 128            // 4 waves of 32
#define Y_STRIDE 68            // padded f32 stride for u state (bank spread)

#define LN2f     0.69314718055994530942f
#define INV_LN2f 1.44269504088896340736f

// raw hardware transcendentals: v_exp_f32 (exp2), v_log_f32 (log2)
__device__ __forceinline__ float hw_exp2(float x) { return __builtin_amdgcn_exp2f(x); }
__device__ __forceinline__ float hw_log2(float x) { return __builtin_amdgcn_logf(x); }

// DPP lane permutes within a 16-lane row (no LDS traffic, pure VALU)
#define DPP_QUAD_XOR1   0xB1   // quad_perm(1,0,3,2)
#define DPP_QUAD_XOR2   0x4E   // quad_perm(2,3,0,1)
#define DPP_HALF_MIRROR 0x141  // row_half_mirror (mirror within 8 lanes)

template <int CTRL>
__device__ __forceinline__ float dpp_movf(float x) {
    int xi = __float_as_int(x);
    int r  = __builtin_amdgcn_update_dpp(xi, xi, CTRL, 0xF, 0xF, true);
    return __int_as_float(r);
}
// reduce over each aligned group of 8 lanes
__device__ __forceinline__ float dpp_max8(float x) {
    x = fmaxf(x, dpp_movf<DPP_QUAD_XOR1>(x));
    x = fmaxf(x, dpp_movf<DPP_QUAD_XOR2>(x));
    x = fmaxf(x, dpp_movf<DPP_HALF_MIRROR>(x));
    return x;
}
__device__ __forceinline__ float dpp_add8(float x) {
    x += dpp_movf<DPP_QUAD_XOR1>(x);
    x += dpp_movf<DPP_QUAD_XOR2>(x);
    x += dpp_movf<DPP_HALF_MIRROR>(x);
    return x;
}

// ---------------------------------------------------------------------------
// Kernel 1: row-softmax each transition matrix w_t (64x64) and store f16,
// pre-swizzled into a B-fragment layout for v_wmma_f32_16x16x32_f16 (k contiguous per lane half):
//   frag(wt, ntile, kblock): 32 lanes x 16 halves, contiguous per lane.
//   value c (0..15) of lane L = T[i][j],  i = ntile*16 + (L%16),
//                                         j = kblock*32 + (L/16)*16 + c
// (B[j][i] = T[i][j] since y_new = P x T^T)
// ---------------------------------------------------------------------------
__global__ __launch_bounds__(64)
void hmm_prep_T(const float* __restrict__ wts, _Float16* __restrict__ swz) {
    const int wt = blockIdx.x;          // 0..510
    const int i  = threadIdx.x;         // transition row 0..63
    const float* row = wts + ((size_t)wt * W + i) * W;

    float mx = -INFINITY;
    for (int j = 0; j < W; ++j) mx = fmaxf(mx, row[j]);
    float su = 0.f;
    for (int j = 0; j < W; ++j) su += hw_exp2((row[j] - mx) * INV_LN2f);
    const float inv = 1.f / su;

    const int nt = i >> 4, nl = i & 15;
    for (int j = 0; j < W; ++j) {
        const int kb = j >> 5;
        const int kh = (j >> 4) & 1;
        const int c  = j & 15;
        const int lane = kh * 16 + nl;
        const size_t idx = (((size_t)wt * 4 + nt) * 2 + kb) * 512 + (size_t)lane * 16 + c;
        swz[idx] = (_Float16)(hw_exp2((row[j] - mx) * INV_LN2f) * inv);
    }
}

// ---------------------------------------------------------------------------
// Kernel 2: forward scan in LOG2 space, deferred emission (u_t = y_t - e_t,
// u_0 = 0), symbols software-pipelined in registers (loaded from global one
// iteration ahead), row reductions via DPP (no LDS bpermute).
//   phase 1: y = u + e_{t-1} (transposed leaf table, contiguous b128 loads),
//            DPP row-max, p = exp2(y - m) -> f16 A-staging
//   phase 2: 2 chained v_wmma_f32_16x16x32_f16 (K=64), u' = m + log2(S)
// Final: add e_{L-1} + final weights, DPP logsumexp2, scale by ln2.
// ---------------------------------------------------------------------------
__global__ __launch_bounds__(THREADS)
void hmm_scan(const int* __restrict__ data,
              const float* __restrict__ distros,
              const float* __restrict__ finalw,
              const _Float16* __restrict__ swz,
              float* __restrict__ out) {
    __shared__ __align__(16) float leafs2T[NBSYM][W]; // log2-softmax, TRANSPOSED
    __shared__ __align__(16) float y[BM][Y_STRIDE];   // running u-state (log2)
    __shared__ __align__(16) _Float16 p[BM][W];       // exp2(y - m), A-matrix staging
    __shared__ __align__(16) float mrow[BM];          // per-row max
    __shared__ __align__(16) float lf2[W];            // final weights / ln2
    __shared__ float log2Zf;                          // log2(sum exp(final weights))

    const int tid  = threadIdx.x;
    const int b0   = blockIdx.x * BM;
    const int wave = tid >> 5;          // 0..3  -> N-tile
    const int lane = tid & 31;
    const int g    = lane >> 4;         // lane half
    const int nl   = lane & 15;
    const int r8   = tid >> 3;          // row for 8-thread-per-row phases
    const int seg  = tid & 7;

    // ---- init: log2-softmax of emission table (transposed), final wts ----
    if (tid < W) {
        const float* rw = distros + tid * NBSYM;
        float mx = -INFINITY;
        for (int s = 0; s < NBSYM; ++s) mx = fmaxf(mx, rw[s]);
        float su = 0.f;
        for (int s = 0; s < NBSYM; ++s) su += hw_exp2((rw[s] - mx) * INV_LN2f);
        const float lz2 = mx * INV_LN2f + hw_log2(su);   // log2 of partition
        for (int s = 0; s < NBSYM; ++s) leafs2T[s][tid] = rw[s] * INV_LN2f - lz2;
        lf2[tid] = finalw[tid] * INV_LN2f;
    }
    if (tid == 0) {                     // log2-partition of final layer
        float mx = -INFINITY;
        for (int i = 0; i < W; ++i) mx = fmaxf(mx, finalw[i]);
        float su = 0.f;
        for (int i = 0; i < W; ++i) su += hw_exp2((finalw[i] - mx) * INV_LN2f);
        log2Zf = mx * INV_LN2f + hw_log2(su);
    }

    // per-thread symbol stream for this thread's row (register pipelined)
    const int* drow = data + (size_t)(b0 + r8) * SEQLEN;
    int s_cur = drow[0];                // symbol at time t-1 for iteration t=1

    // u_0 = 0 (emission e_0 folded into first phase-1)
    {
        #pragma unroll
        for (int k = 0; k < 8; ++k)
            y[r8][seg * 8 + k] = 0.0f;
    }
    __syncthreads();

    // ---- scan over time --------------------------------------------------
    for (int t = 1; t < SEQLEN; ++t) {
        const int wt = t - 1;
        const int s_nxt = drow[t];      // for next iteration (latency hidden)

        // phase 1: y = u + e_{t-1}; DPP row-max over 8 lanes; exp2 -> f16
        const float* eRow = &leafs2T[s_cur][0];
        float v[8];
        float mx = -INFINITY;
        #pragma unroll
        for (int k = 0; k < 8; ++k) {
            v[k] = y[r8][seg * 8 + k] + eRow[seg * 8 + k];
            mx = fmaxf(mx, v[k]);
        }
        mx = dpp_max8(mx);
        #pragma unroll
        for (int k = 0; k < 8; ++k)
            p[r8][seg * 8 + k] = (_Float16)hw_exp2(v[k] - mx);
        if (seg == 0) mrow[r8] = mx;
        __syncthreads();

        // phase 2: WMMA  S = P x T^T  (K = 64 as two chained K=32 WMMAs)
        // A fragments from LDS: lane m = nl, K-halves split by g
        union { v16h vv; v8h h[2]; } a0, a1;
        a0.h[0] = *(const v8h*)&p[nl][     g * 16];
        a0.h[1] = *(const v8h*)&p[nl][ 8 + g * 16];
        a1.h[0] = *(const v8h*)&p[nl][32 + g * 16];
        a1.h[1] = *(const v8h*)&p[nl][40 + g * 16];
        // B fragments from pre-swizzled global: 32 contiguous bytes per lane
        const _Float16* fb = swz + (((size_t)wt * 4 + wave) * 2) * 512;
        const v16h bf0 = *(const v16h*)(fb + (size_t)lane * 16);
        const v16h bf1 = *(const v16h*)(fb + 512 + (size_t)lane * 16);

        v8f acc = {};
        acc = __builtin_amdgcn_wmma_f32_16x16x32_f16(
            false, a0.vv, false, bf0, (short)0, acc, false, false);
        acc = __builtin_amdgcn_wmma_f32_16x16x32_f16(
            false, a1.vv, false, bf1, (short)0, acc, false, false);

        // prefetch next step's B fragment (global_prefetch_b8)
        if (t < NSTEPS) {
            const _Float16* nb = swz + (((size_t)t * 4 + wave) * 2) * 512 + (size_t)lane * 16;
            __builtin_prefetch((const void*)nb, 0, 3);
        }

        // u' = m + log2(S); write back.  D layout: vgpr rr -> M = g*8+rr, N = nl
        #pragma unroll
        for (int rr = 0; rr < 8; ++rr) {
            const int mi = g * 8 + rr;
            const int n  = wave * 16 + nl;
            y[mi][n] = mrow[mi] + hw_log2(acc[rr]);
        }
        s_cur = s_nxt;
        __syncthreads();
    }

    // ---- final: y_L = u + e_{L-1};  out = ln2*(logsumexp2(y_L+lf2)-log2Zf)
    {
        const float* eRow = &leafs2T[s_cur][0];        // s_cur == data[b][511]
        float v[8];
        float mx = -INFINITY;
        #pragma unroll
        for (int k = 0; k < 8; ++k) {
            const int i = seg * 8 + k;
            v[k] = y[r8][i] + eRow[i] + lf2[i];
            mx = fmaxf(mx, v[k]);
        }
        mx = dpp_max8(mx);
        float su = 0.f;
        #pragma unroll
        for (int k = 0; k < 8; ++k) su += hw_exp2(v[k] - mx);
        su = dpp_add8(su);
        if (seg == 0)
            out[b0 + r8] = (mx + hw_log2(su) - log2Zf) * LN2f;
    }
}

// ---------------------------------------------------------------------------
extern "C" void kernel_launch(void* const* d_in, const int* in_sizes, int n_in,
                              void* d_out, int out_size, void* d_ws, size_t ws_size,
                              hipStream_t stream) {
    const int*   data    = (const int*)  d_in[0];   // (B, L) int32
    const float* distros = (const float*)d_in[1];   // (W, NB) f32
    const float* wts     = (const float*)d_in[2];   // (L-1, W, W) f32
    const float* finalw  = (const float*)d_in[3];   // (W,) f32
    _Float16*    swz     = (_Float16*)d_ws;         // 511*4*2*512 halves = 4.1 MB

    hmm_prep_T<<<NSTEPS, 64, 0, stream>>>(wts, swz);
    hmm_scan<<<B_TOT / BM, THREADS, 0, stream>>>(data, distros, finalw, swz,
                                                 (float*)d_out);
}
